// MultiHeadSelfAttention_50319836840450
// MI455X (gfx1250) — matrix operand order repaired and checked by reading, not compiled, not run on hardware
//
#include <hip/hip_runtime.h>
#include <hip/hip_bf16.h>
#include <math.h>

#define MODEL_DIM 1024
#define NUM_HEADS 16
#define HEAD_DIM  64
#define BATCH     2
#define SEQ       2048
#define TOKENS    (BATCH * SEQ)      // 4096
#define QKV_DIM   (3 * MODEL_DIM)    // 3072

typedef _Float16 h16;
typedef __attribute__((ext_vector_type(16))) _Float16 v16h;
typedef __attribute__((ext_vector_type(8)))  _Float16 v8h;
typedef __attribute__((ext_vector_type(8)))  float    v8f;
typedef __attribute__((ext_vector_type(4)))  unsigned int u32x4;
typedef __attribute__((ext_vector_type(8)))  int          i32x8;
typedef __attribute__((ext_vector_type(4)))  int          i32x4;

union V16U { v16h v; v8h h[2]; };

// ---------------------------------------------------------------------------
// Tensor Data Mover: 2D tile load (global -> LDS), D# per CDNA5 ISA.
//   tile_w elems per row (f16), tile_h rows, row stride in elems.
//   pad codes: interval c -> pad every 2^(c+1) DWORDs, amount c -> c+1 DWORDs.
__device__ __forceinline__ void tdm_load_2d(unsigned lds_addr, const void* gptr,
                                            unsigned tile_w, unsigned tile_h,
                                            unsigned stride_elems,
                                            unsigned pad_int_c, unsigned pad_amt_c) {
  const unsigned long long ga = (unsigned long long)(uintptr_t)gptr;
  u32x4 g0;
  g0[0] = 1u;                                            // count=1 (valid user D#)
  g0[1] = lds_addr;                                      // LDS byte address
  g0[2] = (unsigned)(ga & 0xFFFFFFFFu);                  // global_addr[31:0]
  g0[3] = (unsigned)((ga >> 32) & 0x01FFFFFFu) | (2u << 30);   // addr[56:32] | type=2
  i32x8 g1;
  g1[0] = (int)((1u << 16) |                             // data_size = 2 bytes
                (1u << 20) |                             // pad_enable
                (pad_int_c << 22) | (pad_amt_c << 25));
  g1[1] = (int)((tile_w & 0xFFFFu) << 16);               // tensor_dim0[15:0]
  g1[2] = (int)(((tile_w >> 16) & 0xFFFFu) |             // tensor_dim0[31:16]
                ((tile_h & 0xFFFFu) << 16));             // tensor_dim1[15:0]
  g1[3] = (int)(((tile_h >> 16) & 0xFFFFu) |             // tensor_dim1[31:16]
                ((tile_w & 0xFFFFu) << 16));             // tile_dim0
  g1[4] = (int)(tile_h & 0xFFFFu);                       // tile_dim1 (tile_dim2=0)
  g1[5] = (int)stride_elems;                             // tensor_dim0_stride[31:0]
  g1[6] = 0;                                             // stride0[47:32] | stride1 lo
  g1[7] = 0;
  const i32x4 z4 = {0, 0, 0, 0};
#if __clang_major__ >= 23
  const i32x8 z8 = {0, 0, 0, 0, 0, 0, 0, 0};
  __builtin_amdgcn_tensor_load_to_lds(g0, g1, z4, z4, z8, 0);
#else
  __builtin_amdgcn_tensor_load_to_lds(g0, g1, z4, z4, 0);
#endif
}

__device__ __forceinline__ unsigned lds_off(const void* p) {
  return (unsigned)(uintptr_t)p;   // LDS aperture: addr[31:0] = LDS byte offset
}

// A operand, 16x32 f16 tile (row-major, ld in elems). Contiguous K layout:
// lane=row M, lanes0-15 K=0..15, lanes16-31 K=16..31.
__device__ __forceinline__ v16h load_a16x32(const h16* __restrict__ t, int ld) {
  const int lane = threadIdx.x & 31;
  const h16* p = t + (size_t)(lane & 15) * ld + ((lane >> 4) << 4);
  V16U u;
  u.h[0] = *(const v8h*)(p);
  u.h[1] = *(const v8h*)(p + 8);
  return u.v;
}

// B operand, 32(K)x16(N) tile from N-major storage: lane=col N,
// lanes0-15 K=0..15 contiguous, lanes16-31 K=16..31.
__device__ __forceinline__ v16h load_b32x16(const h16* __restrict__ t, int ld) {
  const int lane = threadIdx.x & 31;
  const h16* p = t + (size_t)(lane & 15) * ld + ((lane >> 4) << 4);
  V16U u;
  u.h[0] = *(const v8h*)(p);
  u.h[1] = *(const v8h*)(p + 8);
  return u.v;
}

__device__ __forceinline__ v8f wmma_f16(v16h a, v16h b, v8f c) {
  return __builtin_amdgcn_wmma_f32_16x16x32_f16(false, a, false, b, (short)0, c,
                                                false, false);
}

// ---------------------------------------------------------------------------
__global__ void mha_cast_f16(const float* __restrict__ in, h16* __restrict__ out, int n) {
  int i = blockIdx.x * blockDim.x + threadIdx.x;
  if (i < n) out[i] = (h16)in[i];
}

__global__ void mha_transpose_f16(const float* __restrict__ in, h16* __restrict__ out,
                                  int R, int C) {
  int i = blockIdx.x * blockDim.x + threadIdx.x;
  if (i < R * C) {
    int c = i / R;
    int r = i - c * R;
    out[i] = (h16)in[(size_t)r * C + c];
  }
}

// ---------------------------------------------------------------------------
// WMMA GEMM core: block tile 128(M) x 256(N), 8 waves (2x4), wave tile 64x64.
// Operands TDM-staged in LDS, double buffered, padded rows (32->40 halves).
#define PADW 40
#define NSTEP (MODEL_DIM / 32)

#define GEMM_BODY(Aptr, Wtptr)                                                       \
  __shared__ h16 sA[2][128 * PADW];                                                  \
  __shared__ h16 sB[2][256 * PADW];                                                  \
  const int wave = threadIdx.x >> 5;                                                 \
  const int lane = threadIdx.x & 31;                                                 \
  const int row0 = blockIdx.x * 128 + (wave & 1) * 64;                               \
  const int col0 = blockIdx.y * 256 + (wave >> 1) * 64;                              \
  const h16* gA = (Aptr) + (size_t)blockIdx.x * 128 * MODEL_DIM;                     \
  const h16* gB = (Wtptr) + (size_t)blockIdx.y * 256 * MODEL_DIM;                    \
  if (wave == 0) {                                                                   \
    tdm_load_2d(lds_off(&sA[0][0]), gA, 32, 128, MODEL_DIM, 3, 3);                   \
    tdm_load_2d(lds_off(&sB[0][0]), gB, 32, 256, MODEL_DIM, 3, 3);                   \
  }                                                                                  \
  v8f acc[4][4] = {};                                                                \
  for (int s = 0; s < NSTEP; ++s) {                                                  \
    const int cur = s & 1;                                                           \
    if (wave == 0) __builtin_amdgcn_s_wait_tensorcnt(0);                             \
    __syncthreads(); /* buf[cur] ready; all prior LDS reads retired (dscnt 0) */     \
    if (wave == 0 && s + 1 < NSTEP) {                                                \
      tdm_load_2d(lds_off(&sA[cur ^ 1][0]), gA + (s + 1) * 32, 32, 128, MODEL_DIM,   \
                  3, 3);                                                             \
      tdm_load_2d(lds_off(&sB[cur ^ 1][0]), gB + (s + 1) * 32, 32, 256, MODEL_DIM,   \
                  3, 3);                                                             \
    }                                                                                \
    v16h a[4], b[4];                                                                 \
    _Pragma("unroll") for (int i = 0; i < 4; ++i)                                    \
        a[i] = load_a16x32(&sA[cur][((wave & 1) * 64 + i * 16) * PADW], PADW);       \
    _Pragma("unroll") for (int j = 0; j < 4; ++j)                                    \
        b[j] = load_b32x16(&sB[cur][((wave >> 1) * 64 + j * 16) * PADW], PADW);      \
    _Pragma("unroll") for (int i = 0; i < 4; ++i)                                    \
        _Pragma("unroll") for (int j = 0; j < 4; ++j)                                \
            acc[i][j] = wmma_f16(a[i], b[j], acc[i][j]);                             \
  }

// QKV GEMM + scatter into Q (pre-scaled by 1/sqrt(d)), K, V^T (all f16)
__global__ __launch_bounds__(256) void mha_qkv_gemm(
    const h16* __restrict__ X, const h16* __restrict__ Wt,
    const float* __restrict__ bias,
    h16* __restrict__ Q, h16* __restrict__ K, h16* __restrict__ Vt) {
  GEMM_BODY(X, Wt)
  const int nloc  = lane & 15;
  const int mbase = (lane >> 4) * 8;
#pragma unroll
  for (int j = 0; j < 4; ++j) {
    const int n    = col0 + j * 16 + nloc;   // 0..3071
    const float bv = bias[n];
    const int sect = n >> 10;                // 0:Q 1:K 2:V
    const int f    = n & (MODEL_DIM - 1);
    const int h    = f >> 6;
    const int d    = f & (HEAD_DIM - 1);
#pragma unroll
    for (int i = 0; i < 4; ++i) {
#pragma unroll
      for (int r = 0; r < 8; ++r) {
        const int m   = row0 + i * 16 + mbase + r;   // token index
        const int bi  = m >> 11;
        const int ss  = m & (SEQ - 1);
        const int bh  = bi * NUM_HEADS + h;
        const float v = acc[i][j][r] + bv;
        if (sect == 0)
          Q[((size_t)bh * SEQ + ss) * HEAD_DIM + d] = (h16)(v * 0.125f);
        else if (sect == 1)
          K[((size_t)bh * SEQ + ss) * HEAD_DIM + d] = (h16)v;
        else
          Vt[((size_t)bh * HEAD_DIM + d) * SEQ + ss] = (h16)v;
      }
    }
  }
}

// Output GEMM -> fp32 d_out
__global__ __launch_bounds__(256) void mha_out_gemm(
    const h16* __restrict__ A, const h16* __restrict__ Wt,
    const float* __restrict__ bias, float* __restrict__ out) {
  GEMM_BODY(A, Wt)
  const int nloc  = lane & 15;
  const int mbase = (lane >> 4) * 8;
#pragma unroll
  for (int j = 0; j < 4; ++j) {
    const int n    = col0 + j * 16 + nloc;
    const float bv = bias[n];
#pragma unroll
    for (int i = 0; i < 4; ++i) {
#pragma unroll
      for (int r = 0; r < 8; ++r) {
        const int m = row0 + i * 16 + mbase + r;
        out[(size_t)m * MODEL_DIM + n] = acc[i][j][r] + bv;
      }
    }
  }
}

// ---------------------------------------------------------------------------
// Flash attention. Scores computed TRANSPOSED (M=keys, N=queries): softmax
// stats + ctx rescale are per-lane scalars; P re-packs in-register into the
// PV B operand. K/V chunks (shared by all 8 waves) are TDM-staged in LDS.
#define PADK 72   // 64-half K rows + 8
#define PADV 40   // 32-half V^T rows + 8
#define NCHUNK (SEQ / 32)

__global__ __launch_bounds__(256) void mha_flash_attn(
    const h16* __restrict__ Q, const h16* __restrict__ K,
    const h16* __restrict__ Vt, h16* __restrict__ Ctx) {
  __shared__ h16 sK[2][32 * PADK];
  __shared__ h16 sV[2][64 * PADV];
  const int bh   = blockIdx.x;                 // 0..31
  const int wave = threadIdx.x >> 5;
  const int lane = threadIdx.x & 31;
  const int q0   = blockIdx.y * 128 + wave * 16;

  const h16* Qb = Q  + (size_t)bh * SEQ * HEAD_DIM;
  const h16* Kb = K  + (size_t)bh * SEQ * HEAD_DIM;
  const h16* Vb = Vt + (size_t)bh * HEAD_DIM * SEQ;

  const v16h bq0 = load_b32x16(Qb + (size_t)q0 * HEAD_DIM + 0, HEAD_DIM);
  const v16h bq1 = load_b32x16(Qb + (size_t)q0 * HEAD_DIM + 32, HEAD_DIM);

  if (wave == 0) {
    tdm_load_2d(lds_off(&sK[0][0]), Kb, 64, 32, HEAD_DIM, 4, 3);  // 32 keys x 64d
    tdm_load_2d(lds_off(&sV[0][0]), Vb, 32, 64, SEQ, 3, 3);       // 64d x 32 keys
  }

  v8f ctx[4] = {};
  float m_run = -INFINITY, l_run = 0.f;

  for (int c = 0; c < NCHUNK; ++c) {
    const int cur = c & 1;
    if (wave == 0) __builtin_amdgcn_s_wait_tensorcnt(0);
    __syncthreads();
    if (wave == 0 && c + 1 < NCHUNK) {
      tdm_load_2d(lds_off(&sK[cur ^ 1][0]), Kb + (size_t)(c + 1) * 32 * HEAD_DIM,
                  64, 32, HEAD_DIM, 4, 3);
      tdm_load_2d(lds_off(&sV[cur ^ 1][0]), Vb + (size_t)(c + 1) * 32,
                  32, 64, SEQ, 3, 3);
    }

    // scores^T = K(16x64) . Q^T(64x16), two key subtiles
    v16h a00 = load_a16x32(&sK[cur][0 * PADK + 0], PADK);
    v16h a01 = load_a16x32(&sK[cur][0 * PADK + 32], PADK);
    v16h a10 = load_a16x32(&sK[cur][16 * PADK + 0], PADK);
    v16h a11 = load_a16x32(&sK[cur][16 * PADK + 32], PADK);
    v8f s0 = {}, s1 = {};
    s0 = wmma_f16(a00, bq0, s0);
    s0 = wmma_f16(a01, bq1, s0);
    s1 = wmma_f16(a10, bq0, s1);
    s1 = wmma_f16(a11, bq1, s1);

    // online softmax over 32 keys: 8+8 in-lane + partner half via shfl_xor(16)
    float mloc = s0[0];
#pragma unroll
    for (int r = 1; r < 8; ++r) mloc = fmaxf(mloc, s0[r]);
#pragma unroll
    for (int r = 0; r < 8; ++r) mloc = fmaxf(mloc, s1[r]);
    mloc = fmaxf(mloc, __shfl_xor(mloc, 16, 32));
    const float m_new = fmaxf(m_run, mloc);
    const float alpha = __expf(m_run - m_new);

    float p0[8], p1[8];
    float rs = 0.f;
#pragma unroll
    for (int r = 0; r < 8; ++r) { p0[r] = __expf(s0[r] - m_new); rs += p0[r]; }
#pragma unroll
    for (int r = 0; r < 8; ++r) { p1[r] = __expf(s1[r] - m_new); rs += p1[r]; }
    rs += __shfl_xor(rs, 16, 32);
    l_run = l_run * alpha + rs;
    m_run = m_new;

#pragma unroll
    for (int t = 0; t < 4; ++t)
#pragma unroll
      for (int r = 0; r < 8; ++r) ctx[t][r] *= alpha;

    // Re-pack P (scores^T C-layout) into PV B-operand layout
    V16U bp;
#pragma unroll
    for (int r = 0; r < 8; ++r) {
      const float t0 = __shfl_xor(p0[r], 16, 32);
      const float t1 = __shfl_xor(p1[r], 16, 32);
      const float lo = (lane < 16) ? p0[r] : t1;
      const float hi = (lane < 16) ? t0 : p1[r];
      bp.h[0][r] = (h16)lo;
      bp.h[1][r] = (h16)hi;
    }

    // ctx^T += V^T(16d x 32k) . P(32k x 16q)
#pragma unroll
    for (int t = 0; t < 4; ++t) {
      v16h av = load_a16x32(&sV[cur][(t * 16) * PADV], PADV);
      ctx[t] = wmma_f16(av, bp.v, ctx[t]);
    }
  }

  const float inv_l = 1.f / l_run;
  const int b = bh >> 4, h = bh & 15;
  const int qc = lane & 15, half = lane >> 4;
  const size_t token = (size_t)b * SEQ + q0 + qc;
#pragma unroll
  for (int t = 0; t < 4; ++t) {
    v8h o;
#pragma unroll
    for (int r = 0; r < 8; ++r) o[r] = (h16)(ctx[t][r] * inv_l);
    *(v8h*)(Ctx + token * MODEL_DIM + h * HEAD_DIM + t * 16 + half * 8) = o;
  }
}

// ---------------------------------------------------------------------------
extern "C" void kernel_launch(void* const* d_in, const int* in_sizes, int n_in,
                              void* d_out, int out_size, void* d_ws, size_t ws_size,
                              hipStream_t stream) {
  (void)in_sizes; (void)n_in; (void)out_size; (void)ws_size;
  const float* emb  = (const float*)d_in[0];
  const float* wqkv = (const float*)d_in[1];
  const float* bqkv = (const float*)d_in[2];
  const float* wout = (const float*)d_in[3];
  const float* bout = (const float*)d_in[4];
  float* out = (float*)d_out;

  const size_t N_X  = (size_t)TOKENS * MODEL_DIM;
  const size_t N_W1 = (size_t)QKV_DIM * MODEL_DIM;
  const size_t N_W2 = (size_t)MODEL_DIM * MODEL_DIM;
  const size_t N_HD = (size_t)BATCH * NUM_HEADS * SEQ * HEAD_DIM;

  h16* base = (h16*)d_ws;
  size_t off = 0;
  h16* Xh  = base + off; off += N_X;
  h16* W1t = base + off; off += N_W1;
  h16* W2t = base + off; off += N_W2;
  h16* Qh  = base + off; off += N_HD;
  h16* Kh  = base + off; off += N_HD;
  h16* Vt  = base + off; off += N_HD;
  h16* Ctx = base + off; off += N_X;   // ~50 MB f16 staging total

  mha_cast_f16<<<(int)((N_X + 255) / 256), 256, 0, stream>>>(emb, Xh, (int)N_X);
  mha_transpose_f16<<<(int)((N_W1 + 255) / 256), 256, 0, stream>>>(wqkv, W1t,
      MODEL_DIM, QKV_DIM);
  mha_transpose_f16<<<(int)((N_W2 + 255) / 256), 256, 0, stream>>>(wout, W2t,
      MODEL_DIM, MODEL_DIM);

  mha_qkv_gemm<<<dim3(TOKENS / 128, QKV_DIM / 256), 256, 0, stream>>>(
      Xh, W1t, bqkv, Qh, Kh, Vt);

  mha_flash_attn<<<dim3(BATCH * NUM_HEADS, SEQ / 128), 256, 0, stream>>>(
      Qh, Kh, Vt, Ctx);

  mha_out_gemm<<<dim3(TOKENS / 128, MODEL_DIM / 256), 256, 0, stream>>>(
      Ctx, W2t, bout, out);
}
